// RNNTLoss_44375602103090
// MI455X (gfx1250) — compile-verified, matrix operand order checked by reading, not run on hardware
//
#include <hip/hip_runtime.h>
#include <math.h>

// Problem constants (from reference): B=8, T=256, U=128, V=256, BLANK=0
#define Bsz   8
#define Tlen  256
#define Ulen  128
#define U1    129
#define Vocab 256
#define NROWS (Bsz * Tlen * U1)   // 264192 (b,t,u) rows of length V

typedef __attribute__((ext_vector_type(16))) _Float16 v16h;
typedef __attribute__((ext_vector_type(8)))  float    v8f;

__device__ __forceinline__ float logaddexpf_(float x, float y) {
    float m = fmaxf(x, y);
    if (m == -INFINITY) return -INFINITY;
    return m + log1pf(expf(fminf(x, y) - m));
}

// ---------------------------------------------------------------------------
// Stage 1: streaming log-softmax extraction (HBM-bound: 270MB @ 23.3TB/s).
// For every (b,t,u) row of V=256 logits, compute lse = max + log(sum exp(x-max))
// and emit blk = x[0]-lse and lbl = x[label[b,u]]-lse (u<U) into a 2.1MB
// L2-resident workspace. The exp-row-sum is done with V_WMMA_F32_16X16X32_F16
// against an all-ones B matrix: 16 rows per wave, 8 WMMAs of K=32 each,
// co-executing with the v_exp_f32 stream (16-bit WMMA is TRANS-tracked).
// __expf is used since x-max <= 0 (no overflow path needed) and results are
// truncated to f16 anyway; this collapses ~10 VALU ops/elem to ~2.
//
// A-fragment mapping (ISA 7.12.2, 16-bit A 16x32): lane l holds row M=l&15;
// lanes l<16 hold K={c*32+0..7, c*32+16..23}, lanes l>=16 hold the other half.
// Since B==ones, only the row (M) mapping matters for the sum.
// ---------------------------------------------------------------------------
__global__ void __launch_bounds__(256) rnnt_lse_kernel(
    const float* __restrict__ acts, const int* __restrict__ labels,
    float* __restrict__ blk, float* __restrict__ lbl)
{
    const int wave = threadIdx.x >> 5;
    const int lane = threadIdx.x & 31;
    const int tile = blockIdx.x * 8 + wave;        // 16 rows per wave-tile
    const int r0   = tile * 16;
    const int m    = lane & 15;                    // this lane's matrix row
    const float* rp = acts + (size_t)(r0 + m) * Vocab;
    const int cb   = (lane < 16) ? 0 : 8;          // lane-half column base

    // ---- Pass A: row max (each lane covers its 128 assigned elements) ----
    float mx = -INFINITY;
#pragma unroll
    for (int c = 0; c < 8; ++c) {
        const int col = c * 32 + cb;
        const float4 x0 = *(const float4*)(rp + col);
        const float4 x1 = *(const float4*)(rp + col + 4);
        const float4 x2 = *(const float4*)(rp + col + 16);
        const float4 x3 = *(const float4*)(rp + col + 20);
        mx = fmaxf(mx, fmaxf(fmaxf(fmaxf(x0.x, x0.y), fmaxf(x0.z, x0.w)),
                             fmaxf(fmaxf(x1.x, x1.y), fmaxf(x1.z, x1.w))));
        mx = fmaxf(mx, fmaxf(fmaxf(fmaxf(x2.x, x2.y), fmaxf(x2.z, x2.w)),
                             fmaxf(fmaxf(x3.x, x3.y), fmaxf(x3.z, x3.w))));
    }
    // lanes l and l^16 cover the two halves of the same row -> combine
    mx = fmaxf(mx, __shfl_xor(mx, 16, 32));

    // ---- Pass B: exp + f16 pack + WMMA ones-sum (reloads hit WGP$/L2) ----
    v16h ones;
#pragma unroll
    for (int i = 0; i < 16; ++i) ones[i] = (_Float16)1.0f;
    v8f acc = {0.f, 0.f, 0.f, 0.f, 0.f, 0.f, 0.f, 0.f};
#pragma unroll
    for (int c = 0; c < 8; ++c) {
        const int col = c * 32 + cb;
        const float4 x0 = *(const float4*)(rp + col);
        const float4 x1 = *(const float4*)(rp + col + 4);
        const float4 x2 = *(const float4*)(rp + col + 16);
        const float4 x3 = *(const float4*)(rp + col + 20);
        const float xs[16] = {x0.x, x0.y, x0.z, x0.w, x1.x, x1.y, x1.z, x1.w,
                              x2.x, x2.y, x2.z, x2.w, x3.x, x3.y, x3.z, x3.w};
        v16h a;
#pragma unroll
        for (int i = 0; i < 16; ++i) a[i] = (_Float16)__expf(xs[i] - mx);
        // D = A x ones + C  ->  every column of D holds the row sums
        acc = __builtin_amdgcn_wmma_f32_16x16x32_f16(
                  false, a, false, ones, (short)0, acc, false, false);
    }

    // ---- Readback: D[M][N]: lanes 0-15 hold M=0..7 (vgpr j -> M=j),
    //      lanes 16-31 hold M=8..15 (vgpr j -> M=j+8).
    // Owners: lanes 0..7 report rows 0..7 via acc[lane] (= D[l][l]);
    //         lanes 16..23 report rows 8..15 via acc[lane-16] (= D[l-8][l-16]).
    const bool owner = (lane < 8) || (lane >= 16 && lane < 24);
    const int  orow  = (lane < 16) ? lane : (lane - 8);
    const float rm_o = __shfl(mx, orow & 15, 32);  // lane 'orow' tracked row 'orow'
    const int  cidx  = ((lane < 16) ? lane : (lane - 16)) & 7;
    const float sum  = acc[cidx];

    if (owner) {
        const float lse = rm_o + __logf(sum);      // sum in [1,256], well-conditioned
        const unsigned rr = (unsigned)(r0 + orow);
        const size_t   rb = (size_t)rr * Vocab;
        blk[rr] = acts[rb] - lse;                  // BLANK = 0
        const unsigned u  = rr % U1;
        if (u < Ulen) {
            const unsigned bt = rr / U1;           // b*T + t
            const unsigned bb = bt / Tlen;
            const int lab = labels[bb * Ulen + u];
            lbl[rr] = acts[rb + lab] - lse;
        }
    }
}

// ---------------------------------------------------------------------------
// Stage 2: alpha wavefront DP, one workgroup per batch element.
// alpha[t][u] = logaddexp(alpha[t-1][u]+blk[t-1][u], alpha[t][u-1]+lbl[t][u-1])
// with alpha[0][0]=0 and missing predecessors = -inf (matches the reference's
// cumsum init + cumlogsumexp row recurrence exactly). Anti-diagonal s=t+u,
// double-buffered diagonals in LDS, T+U1-1 = 384 barrier steps. Inputs are
// L2-resident (2.1MB). Accurate expf/log1pf here since DP errors compound.
// ---------------------------------------------------------------------------
__global__ void __launch_bounds__(160) rnnt_dp_kernel(
    const float* __restrict__ blk, const float* __restrict__ lbl,
    const int* __restrict__ act_lens, const int* __restrict__ label_lens,
    float* __restrict__ cost)
{
    __shared__ float dA[U1], dB[U1];
    const int b = blockIdx.x;
    const int u = threadIdx.x;
    const int flen = act_lens[b];
    const int glen = label_lens[b];
    const float* BLK = blk + (size_t)b * Tlen * U1;
    const float* LBL = lbl + (size_t)b * Tlen * U1;

    for (int s = 0; s < Tlen + U1 - 1; ++s) {
        float*       cur  = (s & 1) ? dB : dA;
        const float* prev = (s & 1) ? dA : dB;
        if (u < U1 && u <= s) {
            const int t = s - u;
            if (t < Tlen) {
                float val;
                if (s == 0) {
                    val = 0.f;
                } else {
                    // prev[u]   = alpha[t-1][u], prev[u-1] = alpha[t][u-1]
                    const float up = (t > 0) ? prev[u]     + BLK[(t - 1) * U1 + u] : -INFINITY;
                    const float lf = (u > 0) ? prev[u - 1] + LBL[t * U1 + (u - 1)] : -INFINITY;
                    val = logaddexpf_(up, lf);
                }
                cur[u] = val;
                if (t == flen - 1 && u == glen)
                    cost[b] = -(val + BLK[t * U1 + u]);
            }
        }
        __syncthreads();
    }
}

// ---------------------------------------------------------------------------
// Stage 3: deterministic 8-way sum (no float atomics).
// ---------------------------------------------------------------------------
__global__ void rnnt_sum_kernel(const float* __restrict__ cost,
                                float* __restrict__ out)
{
    if (threadIdx.x == 0 && blockIdx.x == 0) {
        float s = 0.f;
        for (int b = 0; b < Bsz; ++b) s += cost[b];
        out[0] = s;
    }
}

extern "C" void kernel_launch(void* const* d_in, const int* in_sizes, int n_in,
                              void* d_out, int out_size, void* d_ws, size_t ws_size,
                              hipStream_t stream)
{
    (void)in_sizes; (void)n_in; (void)out_size; (void)ws_size;
    const float* acts       = (const float*)d_in[0];
    const int*   labels     = (const int*)d_in[1];
    const int*   act_lens   = (const int*)d_in[2];
    const int*   label_lens = (const int*)d_in[3];

    // Workspace: blk[NROWS] f32, lbl[NROWS] f32, cost[B] f32  (~2.1 MB)
    float* blk  = (float*)d_ws;
    float* lbl  = blk + NROWS;
    float* cost = lbl + NROWS;

    // 264192 rows / 16 rows-per-wave / 8 waves-per-block = 2064 blocks (exact)
    rnnt_lse_kernel<<<NROWS / 16 / 8, 256, 0, stream>>>(acts, labels, blk, lbl);
    rnnt_dp_kernel<<<Bsz, 160, 0, stream>>>(blk, lbl, act_lens, label_lens, cost);
    rnnt_sum_kernel<<<1, 32, 0, stream>>>(cost, (float*)d_out);
}
